// SpatialGate_90271622627576
// MI455X (gfx1250) — matrix operand order repaired and checked by reading, not compiled, not run on hardware
//
#include <hip/hip_runtime.h>
#include <hip/hip_bf16.h>
#include <math.h>
#include <stdint.h>

typedef __attribute__((ext_vector_type(16))) _Float16 v16h;
typedef __attribute__((ext_vector_type(8)))  float    v8f;
typedef __attribute__((ext_vector_type(4)))  float    f32x4;

#define HWDIM 128
#define HW    16384        // 128*128
#define NCH   96
#define NIMG  32
#define NPIX  (NIMG * HW)  // 524288 conv/scale pixels
#define BN_EPS 1e-5f

#define TW 136             // padded tile row width (cols 0..133 = x in [-3,130])

// ---------------- Kernel 0: zero the BN accumulators ----------------
__global__ void k_init(float* __restrict__ acc) {
    if (threadIdx.x < 2) acc[threadIdx.x] = 0.0f;
}

// ---------------- Kernel 1: channel pool (max, mean over C) ----------------
// x (32,96,128,128) -> pooled (32,2,128,128): ch0 = max, ch1 = mean
__global__ __launch_bounds__(256) void k_pool(const float* __restrict__ x,
                                              float* __restrict__ pooled) {
    int t  = blockIdx.x * blockDim.x + threadIdx.x;   // 131072 threads
    int g4 = t * 4;                                   // 4 consecutive pixels
    int n  = g4 >> 14;                                // image
    int p  = g4 & 16383;                              // pixel within image
    const float* base = x + (size_t)n * NCH * HW + p;

    f32x4 mx = {-INFINITY, -INFINITY, -INFINITY, -INFINITY};
    f32x4 sm = {0.f, 0.f, 0.f, 0.f};
    #pragma unroll 4
    for (int c = 0; c < NCH; ++c) {
        f32x4 v = *(const f32x4*)(base + (size_t)c * HW);
        #pragma unroll
        for (int i = 0; i < 4; ++i) {
            mx[i] = fmaxf(mx[i], v[i]);
            sm[i] = sm[i] + v[i];
        }
    }
    const float inv = 1.0f / (float)NCH;
    #pragma unroll
    for (int i = 0; i < 4; ++i) sm[i] *= inv;

    *(f32x4*)(pooled + (size_t)(n * 2 + 0) * HW + p) = mx;
    *(f32x4*)(pooled + (size_t)(n * 2 + 1) * HW + p) = sm;
}

// ---------------- Kernel 2: 7x7 conv (2->1 ch) via WMMA + BN partial stats --
// One block = one output row (n, y). 8 waves * 16 pixels = 128 columns.
// WMMA mapping: M = 16 output x-positions, K = taps reordered so that
//   K = group*8 + dx, group = c*7+dy (14 valid + 2 zero rows), dx in [0,7)
// N = 16-way broadcast of the weight vector.
// With that ordering, all fragment LDS offsets are compile-time constants
// except a +hi (lane-half) term folded into a per-lane base pointer.
__global__ __launch_bounds__(256) void k_conv(const float* __restrict__ pooled,
                                              const float* __restrict__ w,   // 98 f32, OIHW
                                              float* __restrict__ convbuf,   // 524288 f32
                                              float* __restrict__ acc) {     // [sum, sumsq]
    __shared__ _Float16 tileh[16 * TW];  // f16 slab: rows 0..13 = (c,dy), 14..15 = 0
    __shared__ _Float16 wKh[128];        // weights remapped to K order (f16)
    __shared__ float convrow[HWDIM];
    __shared__ float red[HWDIM];
    __shared__ float red2[HWDIM];

    const int bn  = blockIdx.x;          // 0..4095
    const int n   = bn >> 7;
    const int y   = bn & 127;
    const int tid = threadIdx.x;

    // Remapped weights: wKh[K] = w[c*49 + dy*7 + dx] for valid (group,dx), else 0
    if (tid < 128) {
        int group = tid >> 3, dx = tid & 7;
        float val = 0.0f;
        if (group < 14 && dx < 7) {
            int c = group / 7, dy = group % 7;
            val = w[c * 49 + dy * 7 + dx];
        }
        wKh[tid] = (_Float16)val;
    }
    // Input slab, f16, zero-padded in x (cols map to x = j-3) and y; rows>=14 zero
    for (int i = tid; i < 16 * TW; i += 256) {
        int group = i / TW;
        int j     = i - group * TW;
        float val = 0.0f;
        if (group < 14) {
            int c  = group / 7, dy = group - c * 7;
            int yy = y + dy - 3;
            int xx = j - 3;
            if (yy >= 0 && yy < HWDIM && xx >= 0 && xx < HWDIM)
                val = pooled[(size_t)((n * 2 + c) * HWDIM + yy) * HWDIM + xx];
        }
        tileh[i] = (_Float16)val;
    }
    __syncthreads();

    const int lane = tid & 31;
    const int wave = tid >> 5;           // 0..7
    const int hi   = lane >> 4;          // lane half (ISA fragment layouts)
    const int m    = lane & 15;
    const int x0   = wave * 16;

    // Per-lane bases: A's K gets +8*hi  (= +1 group = +TW halves),
    //                 B's K gets +8*hi (direct index into wKh)
    const _Float16* laneA = tileh + hi * TW + (x0 + m);
    const _Float16* laneB = wKh + hi * 8;

    union AU { v16h v; _Float16 e[16]; };
    v8f acc8 = {0.f, 0.f, 0.f, 0.f, 0.f, 0.f, 0.f, 0.f};

    #pragma unroll
    for (int g = 0; g < 4; ++g) {
        AU a, b;
        // B fragment (32x16 f16): lane L half j holds B[K = 32g + (j<8 ? j : j+8) + 8*hi][N]
        #pragma unroll
        for (int j = 0; j < 16; ++j) b.e[j] = laneB[32 * g + (j < 8 ? j : j + 8)];
        // A fragment (16x32 f16): lane L, VGPR vv, half h ->
        //   K = 32g + (vv<4 ? 2vv : 8+2vv) + 8*hi + h
        #pragma unroll
        for (int vv = 0; vv < 8; ++vv) {
            const int K0  = 32 * g + (vv < 4 ? 2 * vv : 8 + 2 * vv); // compile-time
            const int off = (K0 >> 3) * TW + (K0 & 7);               // compile-time
            a.e[2 * vv + 0] = laneA[off + 0];
            a.e[2 * vv + 1] = laneA[off + 1];
        }
        acc8 = __builtin_amdgcn_wmma_f32_16x16x32_f16(
            false, a.v, false, b.v, (short)0, acc8, false, false);
    }

    // D replicated over N; lanes with N==0 (lane 0: M=0..7, lane 16: M=8..15)
    if (m == 0) {
        #pragma unroll
        for (int r = 0; r < 8; ++r)
            convrow[x0 + 8 * hi + r] = acc8[r];
    }
    __syncthreads();

    if (tid < HWDIM) {
        float cv = convrow[tid];
        convbuf[(size_t)bn * HWDIM + tid] = cv;
        red[tid]  = cv;
        red2[tid] = cv * cv;
    }
    __syncthreads();
    for (int s = 64; s > 0; s >>= 1) {
        if (tid < s) { red[tid] += red[tid + s]; red2[tid] += red2[tid + s]; }
        __syncthreads();
    }
    if (tid == 0) {
        atomicAdd(&acc[0], red[0]);
        atomicAdd(&acc[1], red2[0]);
    }
}

// ---------------- Kernel 3: BN finalize + sigmoid + x * scale ----------------
__global__ __launch_bounds__(256) void k_scale(const float* __restrict__ x,
                                               const float* __restrict__ convbuf,
                                               const float* __restrict__ acc,
                                               const float* __restrict__ gamma,
                                               const float* __restrict__ beta,
                                               float* __restrict__ out) {
    int t  = blockIdx.x * blockDim.x + threadIdx.x;   // 131072 threads
    int g4 = t * 4;
    int n  = g4 >> 14;
    int p  = g4 & 16383;

    const float invM = 1.0f / (float)NPIX;
    float mean = acc[0] * invM;
    float var  = fmaxf(acc[1] * invM - mean * mean, 0.0f);
    float istd = rsqrtf(var + BN_EPS);
    float gm   = gamma[0];
    float bt   = beta[0];

    f32x4 cv = *(const f32x4*)(convbuf + (size_t)n * HW + p);
    f32x4 sc;
    #pragma unroll
    for (int i = 0; i < 4; ++i) {
        float z = (cv[i] - mean) * istd * gm + bt;
        sc[i] = 1.0f / (1.0f + __expf(-z));
    }

    const float* xb = x   + (size_t)n * NCH * HW + p;
    float*       ob = out + (size_t)n * NCH * HW + p;
    #pragma unroll 4
    for (int c = 0; c < NCH; ++c) {
        f32x4 xv = *(const f32x4*)(xb + (size_t)c * HW);
        f32x4 ov;
        #pragma unroll
        for (int i = 0; i < 4; ++i) ov[i] = xv[i] * sc[i];
        // output is never re-read: keep it out of the caches
        __builtin_nontemporal_store(ov, (f32x4*)(ob + (size_t)c * HW));
    }
}

extern "C" void kernel_launch(void* const* d_in, const int* in_sizes, int n_in,
                              void* d_out, int out_size, void* d_ws, size_t ws_size,
                              hipStream_t stream) {
    const float* x     = (const float*)d_in[0];
    const float* w     = (const float*)d_in[1];
    const float* gamma = (const float*)d_in[2];
    const float* beta  = (const float*)d_in[3];
    float* out = (float*)d_out;

    float* ws      = (float*)d_ws;
    float* acc     = ws;                             // 2 floats (+pad)
    float* pooled  = ws + 64;                        // 32*2*16384 = 1,048,576 f32
    float* convbuf = pooled + (size_t)NIMG * 2 * HW; // 524,288 f32
    (void)in_sizes; (void)n_in; (void)out_size; (void)ws_size;

    k_init <<<1,    64,  0, stream>>>(acc);
    k_pool <<<512,  256, 0, stream>>>(x, pooled);
    k_conv <<<4096, 256, 0, stream>>>(pooled, w, convbuf, acc);
    k_scale<<<512,  256, 0, stream>>>(x, convbuf, acc, gamma, beta, out);
}
